// GATCritic_43928925503543
// MI455X (gfx1250) — matrix operand order repaired and checked by reading, not compiled, not run on hardware
//
#include <hip/hip_runtime.h>
#include <hip/hip_bf16.h>
#include <stdint.h>

#define N_NODES  4096
#define N_EDGES  32768
#define E_TOT    (N_EDGES + N_NODES)   // self loops appended
#define IN_FEAT  256
#define HID      64
#define IN_HEAD  64
#define OUT_FEAT 128
#define OUT_HEAD 5
#define F1       (IN_HEAD * HID)        // 4096
#define F2       (OUT_HEAD * OUT_FEAT)  // 640
#define NEG_SLOPE 0.2f
#define GAT_EPS   1e-16f

typedef __attribute__((ext_vector_type(8)))  __bf16 v8bf;
typedef __attribute__((ext_vector_type(16))) __bf16 v16bf;
typedef __attribute__((ext_vector_type(8)))  float  v8f;

// ---------------------------------------------------------------- utilities

__global__ void fill_f32(float* __restrict__ p, float v, int n) {
    int t = blockIdx.x * blockDim.x + threadIdx.x;
    if (t < n) p[t] = v;
}

__global__ void cvt_bf16(const float* __restrict__ in, __bf16* __restrict__ out, int n) {
    int t = blockIdx.x * blockDim.x + threadIdx.x;
    if (t < n) out[t] = (__bf16)in[t];
}

// Wt[n*K + k] = (bf16) W[k*N + n]   (store B transposed so WMMA B-fragment
// loads are contiguous per lane)
__global__ void cvt_bf16_t(const float* __restrict__ W, __bf16* __restrict__ Wt,
                           int K, int N, int total) {
    int t = blockIdx.x * blockDim.x + threadIdx.x;
    if (t >= total) return;
    int n = t / K, k = t - n * K;
    Wt[t] = (__bf16)W[(size_t)k * N + n];
}

// ---------------------------------------------------------------- WMMA GEMM
// C[M,N] = A[M,K] * B[K,N], A row-major bf16, Bt = B^T row-major [N,K] bf16.
// 256 threads = 8 wave32; each wave computes a 32x32 C macro-tile as 2x2
// 16x16 WMMA tiles (register tiling: each fragment reused twice).
// Block tile: 128x64 (4x2 wave grid). Fragments double-buffered so the 8
// b128 loads of iteration k+1 overlap the 4 WMMAs of iteration k.

__device__ __forceinline__ v16bf load_a_frag(const __bf16* p) {
    // lane half 0: K 0..7 & 16..23 ; half 1 (ptr pre-offset +8): K 8..15 & 24..31
    v8bf lo = *(const v8bf*)(p);
    v8bf hi = *(const v8bf*)(p + 16);
    return __builtin_shufflevector(lo, hi, 0,1,2,3,4,5,6,7,8,9,10,11,12,13,14,15);
}

__device__ __forceinline__ v16bf load_b_frag(const __bf16* p) {
    // lane half 0: K 0..7 & 16..23 ; half 1 (ptr pre-offset +8): K 8..15 & 24..31
    v8bf lo = *(const v8bf*)(p);
    v8bf hi = *(const v8bf*)(p + 16);
    return __builtin_shufflevector(lo, hi, 0,1,2,3,4,5,6,7,8,9,10,11,12,13,14,15);
}

__global__ __launch_bounds__(256)
void wmma_gemm_bf16(const __bf16* __restrict__ A, const __bf16* __restrict__ Bt,
                    float* __restrict__ C, int M, int N, int K) {
    const int wid  = threadIdx.x >> 5;
    const int lane = threadIdx.x & 31;
    const int half = lane >> 4;     // 0: lanes 0-15, 1: lanes 16-31
    const int l15  = lane & 15;
    const int tileM = blockIdx.x * 128 + (wid & 3) * 32;
    const int tileN = blockIdx.y * 64  + (wid >> 2) * 32;

    const __bf16* ap0 = A  + (size_t)(tileM + l15)      * K + half * 8;
    const __bf16* ap1 = A  + (size_t)(tileM + 16 + l15) * K + half * 8;
    const __bf16* bp0 = Bt + (size_t)(tileN + l15)      * K + half * 8;
    const __bf16* bp1 = Bt + (size_t)(tileN + 16 + l15) * K + half * 8;

    v8f acc00 = {}, acc01 = {}, acc10 = {}, acc11 = {};

    // prologue: fragments for kt = 0
    v16bf a0c = load_a_frag(ap0);
    v16bf a1c = load_a_frag(ap1);
    v16bf b0c = load_b_frag(bp0);
    v16bf b1c = load_b_frag(bp1);

    for (int kt = 0; kt < K; kt += 32) {
        // issue next iteration's loads first (kt wraps to 0 on the final,
        // redundant, iteration so loads stay unconditional and pipelined)
        const int ktn = (kt + 32 < K) ? (kt + 32) : 0;
        v16bf a0n = load_a_frag(ap0 + ktn);
        v16bf a1n = load_a_frag(ap1 + ktn);
        v16bf b0n = load_b_frag(bp0 + ktn);
        v16bf b1n = load_b_frag(bp1 + ktn);
        __builtin_prefetch(ap0 + ktn + 32, 0, 3);   // global_prefetch_b8
        __builtin_prefetch(bp0 + ktn + 32, 0, 3);

        acc00 = __builtin_amdgcn_wmma_f32_16x16x32_bf16(false, a0c, false, b0c, (short)0, acc00, false, false);
        acc01 = __builtin_amdgcn_wmma_f32_16x16x32_bf16(false, a0c, false, b1c, (short)0, acc01, false, false);
        acc10 = __builtin_amdgcn_wmma_f32_16x16x32_bf16(false, a1c, false, b0c, (short)0, acc10, false, false);
        acc11 = __builtin_amdgcn_wmma_f32_16x16x32_bf16(false, a1c, false, b1c, (short)0, acc11, false, false);

        a0c = a0n; a1c = a1n; b0c = b0n; b1c = b1n;
    }

    // C/D layout: VGPR j -> row (+half*8 + j), col (+l15)
    float* cp00 = C + (size_t)(tileM + half * 8)      * N + tileN + l15;
    float* cp10 = C + (size_t)(tileM + 16 + half * 8) * N + tileN + l15;
#pragma unroll
    for (int j = 0; j < 8; ++j) {
        cp00[(size_t)j * N]      = acc00[j];
        cp00[(size_t)j * N + 16] = acc01[j];
        cp10[(size_t)j * N]      = acc10[j];
        cp10[(size_t)j * N + 16] = acc11[j];
    }
}

// ---------------------------------------------------------------- attention

// per-(node, head) dot products with att vectors
__global__ void att_scores(const float* __restrict__ hf, const float* __restrict__ aS,
                           const float* __restrict__ aD, int H, int C,
                           float* __restrict__ oS, float* __restrict__ oD) {
    int t = blockIdx.x * blockDim.x + threadIdx.x;
    if (t >= N_NODES * H) return;
    int n = t / H, h = t - n * H;
    const float* hp = hf + ((size_t)n * H + h) * C;
    const float* sp = aS + (size_t)h * C;
    const float* dp = aD + (size_t)h * C;
    float s = 0.f, d = 0.f;
    for (int c = 0; c < C; ++c) { float v = hp[c]; s += v * sp[c]; d += v * dp[c]; }
    oS[t] = s; oD[t] = d;
}

__device__ __forceinline__ void get_edge(const long long* __restrict__ ei,
                                         int e, int& s, int& d) {
    if (e < N_EDGES) { s = (int)ei[e]; d = (int)ei[N_EDGES + e]; }
    else             { s = d = e - N_EDGES; }   // self loop
}

__device__ __forceinline__ void atomicMaxF(float* addr, float val) {
    int old = __float_as_int(*(volatile float*)addr);
    while (__int_as_float(old) < val) {
        int prev = atomicCAS((int*)addr, old, __float_as_int(val));
        if (prev == old) break;
        old = prev;
    }
}

__global__ void edge_max(const float* __restrict__ aS, const float* __restrict__ aD,
                         const long long* __restrict__ ei, int H,
                         float* __restrict__ ev, float* __restrict__ mmax) {
    int t = blockIdx.x * blockDim.x + threadIdx.x;
    if (t >= E_TOT * H) return;
    int e = t / H, h = t - e * H;
    int s, d; get_edge(ei, e, s, d);
    float v = aS[s * H + h] + aD[d * H + h];
    v = v > 0.f ? v : NEG_SLOPE * v;            // leaky_relu
    ev[t] = v;
    atomicMaxF(&mmax[d * H + h], v);
}

__global__ void edge_expsum(float* __restrict__ ev, const float* __restrict__ mmax,
                            const long long* __restrict__ ei, int H,
                            float* __restrict__ den) {
    int t = blockIdx.x * blockDim.x + threadIdx.x;
    if (t >= E_TOT * H) return;
    int e = t / H, h = t - e * H;
    int s, d; get_edge(ei, e, s, d); (void)s;
    float ex = __expf(ev[t] - mmax[d * H + h]);
    ev[t] = ex;                                  // in-place: ev now holds exp
    atomicAdd(&den[d * H + h], ex);
}

// out[dst, h, c] += alpha(e,h) * hf[src, h, c], 4 channels per thread
__global__ void edge_aggregate(const float* __restrict__ hf, const float* __restrict__ ev,
                               const float* __restrict__ den, const long long* __restrict__ ei,
                               int H, int C, float* __restrict__ out) {
    int t = blockIdx.x * blockDim.x + threadIdx.x;
    int c4 = C >> 2;
    int per_edge = H * c4;
    if (t >= E_TOT * per_edge) return;
    int e = t / per_edge;
    int r = t - e * per_edge;
    int h = r / c4;
    int cc = (r - h * c4) << 2;
    int s, d; get_edge(ei, e, s, d);
    float alpha = ev[e * H + h] / (den[d * H + h] + GAT_EPS);
    const float4 v = *(const float4*)(hf + ((size_t)s * H + h) * C + cc);
    float* op = out + ((size_t)d * H + h) * C + cc;
    atomicAdd(op + 0, alpha * v.x);
    atomicAdd(op + 1, alpha * v.y);
    atomicAdd(op + 2, alpha * v.z);
    atomicAdd(op + 3, alpha * v.w);
}

// out_bf16 = bf16(elu(agg + bias[f]))
__global__ void bias_elu_bf16(const float* __restrict__ agg, const float* __restrict__ b,
                              __bf16* __restrict__ out, int F, int total) {
    int t = blockIdx.x * blockDim.x + threadIdx.x;
    if (t >= total) return;
    int f = t % F;
    float v = agg[t] + b[f];
    v = v > 0.f ? v : (__expf(v) - 1.f);
    out[t] = (__bf16)v;
}

// out[c] = tanh( mean_{n,h} agg2[n, h*OUT_FEAT + c] + b2[c] )
__global__ void final_reduce(const float* __restrict__ agg2, const float* __restrict__ b2,
                             float* __restrict__ out) {
    __shared__ float sm[256];
    int c = blockIdx.x;
    float s = 0.f;
    for (int i = threadIdx.x; i < N_NODES * OUT_HEAD; i += 256) {
        int n = i / OUT_HEAD, h = i - n * OUT_HEAD;
        s += agg2[(size_t)n * F2 + h * OUT_FEAT + c];
    }
    sm[threadIdx.x] = s;
    __syncthreads();
    for (int st = 128; st > 0; st >>= 1) {
        if (threadIdx.x < st) sm[threadIdx.x] += sm[threadIdx.x + st];
        __syncthreads();
    }
    if (threadIdx.x == 0)
        out[c] = tanhf(sm[0] / (float)(N_NODES * OUT_HEAD) + b2[c]);
}

// ---------------------------------------------------------------- launcher

static inline int blks(long long n, int t) { return (int)((n + t - 1) / t); }

extern "C" void kernel_launch(void* const* d_in, const int* in_sizes, int n_in,
                              void* d_out, int out_size, void* d_ws, size_t ws_size,
                              hipStream_t stream) {
    (void)in_sizes; (void)n_in; (void)out_size; (void)ws_size;
    const float*     x     = (const float*)d_in[0];
    const long long* ei    = (const long long*)d_in[1];   // int64 [2, E]
    const float*     W1    = (const float*)d_in[2];
    const float*     attS1 = (const float*)d_in[3];
    const float*     attD1 = (const float*)d_in[4];
    const float*     b1    = (const float*)d_in[5];
    const float*     W2    = (const float*)d_in[6];
    const float*     attS2 = (const float*)d_in[7];
    const float*     attD2 = (const float*)d_in[8];
    const float*     b2    = (const float*)d_in[9];
    float*           out   = (float*)d_out;

    // workspace carve-out (256B aligned slabs)
    size_t o = 0;
    char* base = (char*)d_ws;
    auto take = [&](size_t bytes) -> void* {
        void* p = base + o;
        o = (o + bytes + 255) & ~(size_t)255;
        return p;
    };
    __bf16* xb   = (__bf16*)take((size_t)N_NODES * IN_FEAT * 2);
    __bf16* W1t  = (__bf16*)take((size_t)F1 * IN_FEAT * 2);       // [F1][IN_FEAT]
    float*  h1   = (float*) take((size_t)N_NODES * F1 * 4);
    float*  as1  = (float*) take((size_t)N_NODES * IN_HEAD * 4);
    float*  ad1  = (float*) take((size_t)N_NODES * IN_HEAD * 4);
    float*  m1   = (float*) take((size_t)N_NODES * IN_HEAD * 4);
    float*  dn1  = (float*) take((size_t)N_NODES * IN_HEAD * 4);
    float*  ev1  = (float*) take((size_t)E_TOT * IN_HEAD * 4);
    float*  agg1 = (float*) take((size_t)N_NODES * F1 * 4);
    __bf16* h1b  = (__bf16*)take((size_t)N_NODES * F1 * 2);
    __bf16* W2t  = (__bf16*)take((size_t)F2 * F1 * 2);            // [F2][F1]
    float*  h2   = (float*) take((size_t)N_NODES * F2 * 4);
    float*  as2  = (float*) take((size_t)N_NODES * OUT_HEAD * 4);
    float*  ad2  = (float*) take((size_t)N_NODES * OUT_HEAD * 4);
    float*  m2   = (float*) take((size_t)N_NODES * OUT_HEAD * 4);
    float*  dn2  = (float*) take((size_t)N_NODES * OUT_HEAD * 4);
    float*  ev2  = (float*) take((size_t)E_TOT * OUT_HEAD * 4);
    float*  agg2 = (float*) take((size_t)N_NODES * F2 * 4);

    const int T = 256;
    const float NEG_BIG = -3.402823466e38f;

    // ---- layer 1 ----
    cvt_bf16  <<<blks((long long)N_NODES * IN_FEAT, T), T, 0, stream>>>(x, xb, N_NODES * IN_FEAT);
    cvt_bf16_t<<<blks((long long)F1 * IN_FEAT, T), T, 0, stream>>>(W1, W1t, IN_FEAT, F1, F1 * IN_FEAT);
    {
        dim3 g(N_NODES / 128, F1 / 64);
        wmma_gemm_bf16<<<g, T, 0, stream>>>(xb, W1t, h1, N_NODES, F1, IN_FEAT);
    }
    att_scores<<<blks((long long)N_NODES * IN_HEAD, T), T, 0, stream>>>(h1, attS1, attD1, IN_HEAD, HID, as1, ad1);
    fill_f32<<<blks((long long)N_NODES * IN_HEAD, T), T, 0, stream>>>(m1, NEG_BIG, N_NODES * IN_HEAD);
    fill_f32<<<blks((long long)N_NODES * IN_HEAD, T), T, 0, stream>>>(dn1, 0.f, N_NODES * IN_HEAD);
    fill_f32<<<blks((long long)N_NODES * F1, T), T, 0, stream>>>(agg1, 0.f, N_NODES * F1);
    edge_max   <<<blks((long long)E_TOT * IN_HEAD, T), T, 0, stream>>>(as1, ad1, ei, IN_HEAD, ev1, m1);
    edge_expsum<<<blks((long long)E_TOT * IN_HEAD, T), T, 0, stream>>>(ev1, m1, ei, IN_HEAD, dn1);
    edge_aggregate<<<blks((long long)E_TOT * IN_HEAD * (HID / 4), T), T, 0, stream>>>(
        h1, ev1, dn1, ei, IN_HEAD, HID, agg1);
    bias_elu_bf16<<<blks((long long)N_NODES * F1, T), T, 0, stream>>>(agg1, b1, h1b, F1, N_NODES * F1);

    // ---- layer 2 ----
    cvt_bf16_t<<<blks((long long)F2 * F1, T), T, 0, stream>>>(W2, W2t, F1, F2, F2 * F1);
    {
        dim3 g(N_NODES / 128, F2 / 64);
        wmma_gemm_bf16<<<g, T, 0, stream>>>(h1b, W2t, h2, N_NODES, F2, F1);
    }
    att_scores<<<blks((long long)N_NODES * OUT_HEAD, T), T, 0, stream>>>(h2, attS2, attD2, OUT_HEAD, OUT_FEAT, as2, ad2);
    fill_f32<<<blks((long long)N_NODES * OUT_HEAD, T), T, 0, stream>>>(m2, NEG_BIG, N_NODES * OUT_HEAD);
    fill_f32<<<blks((long long)N_NODES * OUT_HEAD, T), T, 0, stream>>>(dn2, 0.f, N_NODES * OUT_HEAD);
    fill_f32<<<blks((long long)N_NODES * F2, T), T, 0, stream>>>(agg2, 0.f, N_NODES * F2);
    edge_max   <<<blks((long long)E_TOT * OUT_HEAD, T), T, 0, stream>>>(as2, ad2, ei, OUT_HEAD, ev2, m2);
    edge_expsum<<<blks((long long)E_TOT * OUT_HEAD, T), T, 0, stream>>>(ev2, m2, ei, OUT_HEAD, dn2);
    edge_aggregate<<<blks((long long)E_TOT * OUT_HEAD * (OUT_FEAT / 4), T), T, 0, stream>>>(
        h2, ev2, dn2, ei, OUT_HEAD, OUT_FEAT, agg2);

    // ---- head-mean + node-mean + bias + tanh ----
    final_reduce<<<OUT_FEAT, 256, 0, stream>>>(agg2, b2, out);
}
